// Enhance_26319559590732
// MI455X (gfx1250) — matrix operand order repaired and checked by reading, not compiled, not run on hardware
//
#include <hip/hip_runtime.h>
#include <hip/hip_bf16.h>

typedef __attribute__((ext_vector_type(16))) _Float16 v16h;
typedef __attribute__((ext_vector_type(8)))  _Float16 v8h;
typedef __attribute__((ext_vector_type(8)))  float    v8f;

#define D_MODELC 128
#define D_INNERC 256
#define NBATCH   8
#define LSEQ     4096
#define NBIG     288                 // [0:16)=Bmat [16:32)=Cmat [32:288)=dt_pre
#define MROWS    (NBATCH * LSEQ)     // 32768
#define SCHUNK   32                  // scan steps staged per async chunk
#define NCHUNK   (LSEQ / SCHUNK)     // 128

static __device__ __forceinline__ float sigmoidf_(float x) { return 1.f / (1.f + __expf(-x)); }
static __device__ __forceinline__ float siluf_(float x)    { return x * sigmoidf_(x); }
static __device__ __forceinline__ float softplusf_(float x){ return (x > 20.f) ? x : log1pf(__expf(x)); }

// ---------------- pack / prep kernels ----------------

// x_hsi (B,C,L) f32 -> xT (B,L,C) f16   (tid == flat index of x_hsi, l fastest)
__global__ __launch_bounds__(256) void pack_x_kernel(const float* __restrict__ x_hsi,
                                                     _Float16* __restrict__ xT) {
  int tid = blockIdx.x * 256 + threadIdx.x;
  int l = tid & (LSEQ - 1);
  int c = (tid >> 12) & 127;
  int b = tid >> 19;
  xT[((size_t)(b * LSEQ + l)) * D_MODELC + c] = (_Float16)x_hsi[tid];
}

// W_in (128,512) f32 -> WinT (512,128) f16 (row-major N x K)
__global__ __launch_bounds__(256) void pack_winT_kernel(const float* __restrict__ W,
                                                        _Float16* __restrict__ WT) {
  int tid = blockIdx.x * 256 + threadIdx.x;  // n*128 + k
  int k = tid & 127, n = tid >> 7;
  WT[tid] = (_Float16)W[k * 512 + n];
}

// W_out (256,128) f32 -> WoutT (128,256) f16
__global__ __launch_bounds__(256) void pack_woutT_kernel(const float* __restrict__ W,
                                                         _Float16* __restrict__ WT) {
  int tid = blockIdx.x * 256 + threadIdx.x;  // n*256 + k
  int k = tid & 255, n = tid >> 8;
  WT[tid] = (_Float16)W[k * 128 + n];
}

// WbigT (288,256): rows 0..15 = W_x B-cols, 16..31 = W_x C-cols,
// rows 32..287 = (W_x[:, :8] @ W_dt)^T  -- folds the dt_low@W_dt GEMM in.
__global__ __launch_bounds__(256) void build_wbigT_kernel(const float* __restrict__ W_x,
                                                          const float* __restrict__ W_dt,
                                                          _Float16* __restrict__ WbigT) {
  int tid = blockIdx.x * 256 + threadIdx.x;  // j*256 + k
  int k = tid & 255;
  int j = tid >> 8;
  if (j >= NBIG) return;
  float v;
  if (j < 16) {
    v = W_x[k * 40 + 8 + j];
  } else if (j < 32) {
    v = W_x[k * 40 + 24 + (j - 16)];
  } else {
    int jo = j - 32;
    v = 0.f;
#pragma unroll
    for (int r = 0; r < 8; ++r) v += W_x[k * 40 + r] * W_dt[r * 256 + jo];
  }
  WbigT[(size_t)j * 256 + k] = (_Float16)v;
}

// ---------------- fragment loaders (ISA 16-bit A/B layouts) ----------------
static __device__ __forceinline__ v16h ld_frag_a(const _Float16* p) {
  // lane = M row; K split {kb..kb+7, 16+kb..16+kb+7}
  v8h lo = *(const v8h*)p;
  v8h hi = *(const v8h*)(p + 16);
  v16h r;
#pragma unroll
  for (int i = 0; i < 8; ++i) { r[i] = lo[i]; r[8 + i] = hi[i]; }
  return r;
}
static __device__ __forceinline__ v16h ld_frag_b(const _Float16* p) {
  // lane = N col; K split {kb..kb+7, 16+kb..16+kb+7}
  v8h lo = *(const v8h*)p;
  v8h hi = *(const v8h*)(p + 16);
  v16h r;
#pragma unroll
  for (int i = 0; i < 8; ++i) { r[i] = lo[i]; r[8 + i] = hi[i]; }
  return r;
}
static __device__ __forceinline__ v8f wmma_(v16h a, v16h b, v8f c) {
  return __builtin_amdgcn_wmma_f32_16x16x32_f16(false, a, false, b, (short)0, c,
                                                false, false);
}

// ---------------- WMMA GEMM, 2x2 tile blocked: C(MxN,f32) = A(MxK) * BT(NxK)^T --------
__global__ __launch_bounds__(256) void gemm_f16_wmma(const _Float16* __restrict__ A,
                                                     const _Float16* __restrict__ BT,
                                                     float* __restrict__ C,
                                                     int M, int N, int K) {
  int gtid = blockIdx.x * 256 + threadIdx.x;
  int wave = gtid >> 5;
  int lane = threadIdx.x & 31;
  int ntile2 = N >> 5;
  int total = (M >> 5) * ntile2;
  if (wave >= total) return;
  int tm = wave / ntile2;
  int tn = wave - tm * ntile2;

  const _Float16* ap0 = A  + ((size_t)tm * 32 + (lane & 15)) * K + ((lane >> 4) * 8);
  const _Float16* ap1 = ap0 + (size_t)16 * K;
  const _Float16* bp0 = BT + ((size_t)tn * 32 + (lane & 15)) * K + ((lane >> 4) * 8);
  const _Float16* bp1 = bp0 + (size_t)16 * K;

  v8f acc00 = {}, acc01 = {}, acc10 = {}, acc11 = {};
  for (int k = 0; k < K; k += 32) {
    v16h a0 = ld_frag_a(ap0 + k);
    v16h a1 = ld_frag_a(ap1 + k);
    v16h b0 = ld_frag_b(bp0 + k);
    v16h b1 = ld_frag_b(bp1 + k);
    acc00 = wmma_(a0, b0, acc00);
    acc01 = wmma_(a0, b1, acc01);
    acc10 = wmma_(a1, b0, acc10);
    acc11 = wmma_(a1, b1, acc11);
  }
  // C/D layout: lanes 0-15 M=r, lanes 16-31 M=8+r; N = lane&15 (coalesced stores)
  int mbase = tm * 32 + (lane >> 4) * 8;
  int nbase = tn * 32 + (lane & 15);
  float* cp00 = C + (size_t)mbase * N + nbase;
#pragma unroll
  for (int r = 0; r < 8; ++r) {
    cp00[(size_t)r * N]            = acc00[r];
    cp00[(size_t)r * N + 16]       = acc01[r];
    cp00[(size_t)(r + 16) * N]     = acc10[r];
    cp00[(size_t)(r + 16) * N + 16]= acc11[r];
  }
}

// ---------------- causal conv(k=2) + SiLU, f32 in -> f16 out ----------------
__global__ __launch_bounds__(256) void conv_silu_kernel(const float* __restrict__ xz,
                                                        const float* __restrict__ conv_w,
                                                        const float* __restrict__ conv_b,
                                                        _Float16* __restrict__ xc) {
  int tid = blockIdx.x * 256 + threadIdx.x;  // (b,l,d), d fastest
  int d = tid & 255;
  int l = (tid >> 8) & (LSEQ - 1);
  int bl = tid >> 8;  // b*L + l
  float cur = xz[(size_t)bl * 512 + d];
  float prev = (l == 0) ? 0.f : xz[(size_t)(bl - 1) * 512 + d];
  float v = prev * conv_w[d * 2 + 0] + cur * conv_w[d * 2 + 1] + conv_b[d];
  xc[(size_t)bl * 256 + d] = (_Float16)siluf_(v);
}

// ---------------- selective scan with async global->LDS double buffering ----------------
// Per wave: stage SCHUNK steps of the 32-float B|C row prefix into LDS via
// GLOBAL_LOAD_ASYNC_TO_LDS_B128 (ASYNCcnt), one chunk ahead of compute.
static __device__ __forceinline__ void issue_chunk(const float* rowbase, int step0,
                                                   unsigned lds_byte, int lane) {
  // 8 async b128 ops; each: 32 lanes x 16B = 512B = 4 steps x 128B (row[0..31])
  const float* gp = rowbase + (size_t)(step0 + (lane >> 3)) * NBIG + (lane & 7) * 4;
  unsigned lp = lds_byte + (unsigned)((lane >> 3) * 128 + (lane & 7) * 16);
#pragma unroll
  for (int j = 0; j < 8; ++j) {
    asm volatile("global_load_async_to_lds_b128 %0, %1, off"
                 :: "v"(lp), "v"(gp) : "memory");
    gp += 4 * NBIG;   // 4 steps forward
    lp += 4 * 128;
  }
}

__global__ __launch_bounds__(256) void scan_kernel(const float* __restrict__ xdbl,
                                                   const _Float16* __restrict__ xc,
                                                   const float* __restrict__ xz,
                                                   const float* __restrict__ A_log,
                                                   const float* __restrict__ b_dt,
                                                   const float* __restrict__ Dvec,
                                                   _Float16* __restrict__ yh) {
  __shared__ float sbuf[8][2][SCHUNK * 32];   // 64 KB: per-wave double buffer
  int lane = threadIdx.x & 31;
  int waveId = threadIdx.x >> 5;
  int dg = blockIdx.x & 15;
  int b  = blockIdx.x >> 4;
  int n = lane & 15;                                // state index
  int d = dg * 16 + waveId * 2 + (lane >> 4);       // channel

  float An  = -__expf(A_log[d * 16 + n]);           // A = -exp(A_log)
  float bdt = b_dt[d];
  float Dd  = Dvec[d];

  const float*    rowb = xdbl + (size_t)b * LSEQ * NBIG;
  const float*    drow = rowb + 32 + d;                      // dt_pre scalar
  const _Float16* xrow = xc   + (size_t)b * LSEQ * 256 + d;
  const float*    zrow = xz   + (size_t)b * LSEQ * 512 + 256 + d;
  _Float16*       yrow = yh   + (size_t)b * LSEQ * 256 + d;

  unsigned lds0 = (unsigned)(uintptr_t)&sbuf[waveId][0][0];  // LDS byte offset
  issue_chunk(rowb, 0, lds0, lane);

  float h = 0.f;
  for (int c = 0; c < NCHUNK; ++c) {
    if (c + 1 < NCHUNK) {
      issue_chunk(rowb, (c + 1) * SCHUNK, lds0 + (((c + 1) & 1) << 12), lane);
      asm volatile("s_wait_asynccnt 0x8" ::: "memory");   // chunk c landed (in-order)
    } else {
      asm volatile("s_wait_asynccnt 0x0" ::: "memory");
    }
    const float* sb = &sbuf[waveId][c & 1][0];
    for (int s = 0; s < SCHUNK; ++s) {
      float Bv  = sb[s * 32 + n];
      float Cv  = sb[s * 32 + 16 + n];
      float dtp = *drow;
      float x   = (float)(*xrow);
      float zv  = *zrow;

      float dt = softplusf_(dtp + bdt);
      float dA = __expf(dt * An);
      h = fmaf(dA, h, dt * Bv * x);

      float p = h * Cv;                               // reduce over 16 states
      p += __shfl_xor(p, 8, 16);
      p += __shfl_xor(p, 4, 16);
      p += __shfl_xor(p, 2, 16);
      p += __shfl_xor(p, 1, 16);

      if (n == 0) {
        float outv = (p + x * Dd) * siluf_(zv);       // +x*D and z-gate fused
        *yrow = (_Float16)outv;
      }
      drow += NBIG; xrow += 256; zrow += 512; yrow += 256;
    }
  }
}

// ---------------- group-norm stats: one block per (b, group) ----------------
__global__ __launch_bounds__(256) void gn_stats_kernel(const float* __restrict__ opre,
                                                       float* __restrict__ stats) {
  __shared__ float s1[256], s2[256];
  int bg = blockIdx.x;  // b*4 + g
  int b = bg >> 2, g = bg & 3;
  float s = 0.f, sq = 0.f;
  for (int i = threadIdx.x; i < 32 * LSEQ; i += 256) {
    int c = g * 32 + (i & 31);
    int l = i >> 5;
    float v = opre[((size_t)(b * LSEQ + l)) * D_MODELC + c];
    s += v; sq += v * v;
  }
  s1[threadIdx.x] = s; s2[threadIdx.x] = sq;
  __syncthreads();
  for (int o = 128; o > 0; o >>= 1) {
    if (threadIdx.x < o) {
      s1[threadIdx.x] += s1[threadIdx.x + o];
      s2[threadIdx.x] += s2[threadIdx.x + o];
    }
    __syncthreads();
  }
  if (threadIdx.x == 0) {
    float inv = 1.f / (32.f * LSEQ);
    float mean = s1[0] * inv;
    float var = s2[0] * inv - mean * mean;
    stats[bg * 2 + 0] = mean;
    stats[bg * 2 + 1] = rsqrtf(var + 1e-5f);
  }
}

// ---------------- normalize + SiLU + residual ----------------
__global__ __launch_bounds__(256) void finalize_kernel(const float* __restrict__ opre,
                                                       const float* __restrict__ x_hsi,
                                                       const float* __restrict__ stats,
                                                       const float* __restrict__ gamma,
                                                       const float* __restrict__ beta,
                                                       float* __restrict__ out) {
  int tid = blockIdx.x * 256 + threadIdx.x;  // (b,c,l), l fastest == output flat index
  int l = tid & (LSEQ - 1);
  int c = (tid >> 12) & 127;
  int b = tid >> 19;
  int g = c >> 5;
  float mean = stats[(b * 4 + g) * 2 + 0];
  float rstd = stats[(b * 4 + g) * 2 + 1];
  float v = opre[((size_t)(b * LSEQ + l)) * D_MODELC + c];
  float y = (v - mean) * rstd * gamma[c] + beta[c];
  out[tid] = siluf_(y) + x_hsi[tid];
}

// ---------------- launch ----------------
extern "C" void kernel_launch(void* const* d_in, const int* in_sizes, int n_in,
                              void* d_out, int out_size, void* d_ws, size_t ws_size,
                              hipStream_t stream) {
  (void)in_sizes; (void)n_in; (void)out_size; (void)ws_size;
  const float* x_hsi  = (const float*)d_in[0];
  const float* W_in   = (const float*)d_in[1];
  const float* conv_w = (const float*)d_in[2];
  const float* conv_b = (const float*)d_in[3];
  const float* W_x    = (const float*)d_in[4];
  const float* W_dt   = (const float*)d_in[5];
  const float* b_dt   = (const float*)d_in[6];
  const float* A_log  = (const float*)d_in[7];
  const float* Dvec   = (const float*)d_in[8];
  const float* W_out  = (const float*)d_in[9];
  const float* gamma  = (const float*)d_in[10];
  const float* beta   = (const float*)d_in[11];
  float* out = (float*)d_out;

  char* ws = (char*)d_ws;                         // all offsets 256B-aligned
  _Float16* xT    = (_Float16*)(ws + 0);          //  8,388,608  (B,L,128) f16
  _Float16* WinT  = (_Float16*)(ws + 8388608);    //    131,072  (512,128) f16
  float*    xz    = (float*)   (ws + 8519680);    // 67,108,864  (B,L,512) f32
  _Float16* xc16  = (_Float16*)(ws + 75628544);   // 16,777,216  (B,L,256) f16
  _Float16* WbigT = (_Float16*)(ws + 92405760);   //    147,456  (288,256) f16
  float*    xdbl  = (float*)   (ws + 92553216);   // 37,748,736  (B,L,288) f32
  _Float16* yh    = (_Float16*)(ws + 130301952);  // 16,777,216  (B,L,256) f16
  _Float16* WoutT = (_Float16*)(ws + 147079168);  //     65,536  (128,256) f16
  float*    opre  = (float*)   (ws + 8519680);    // alias xz (xz dead after scan)
  float*    stats = (float*)   (ws + 147144704);  //        256

  pack_x_kernel    <<<16384, 256, 0, stream>>>(x_hsi, xT);
  pack_winT_kernel <<<  256, 256, 0, stream>>>(W_in, WinT);
  build_wbigT_kernel<<< 288, 256, 0, stream>>>(W_x, W_dt, WbigT);
  pack_woutT_kernel<<<  128, 256, 0, stream>>>(W_out, WoutT);

  // xz = xT @ W_in : (32768,128)x(128,512)   -> 1024x16 = 16384 waves
  gemm_f16_wmma    <<< 2048, 256, 0, stream>>>(xT, WinT, xz, MROWS, 512, 128);
  // conv(k=2)+SiLU over L per channel
  conv_silu_kernel <<<32768, 256, 0, stream>>>(xz, conv_w, conv_b, xc16);
  // xdbl = xc @ [B|C|W_x_dt] : (32768,256)x(256,288) -> 1024x9 = 9216 waves
  gemm_f16_wmma    <<< 1152, 256, 0, stream>>>(xc16, WbigT, xdbl, MROWS, NBIG, 256);
  // sequential selective scan + D-skip + z-gate (async LDS staged)
  scan_kernel      <<<  128, 256, 0, stream>>>(xdbl, xc16, xz, A_log, b_dt, Dvec, yh);
  // out_pre = y @ W_out : (32768,256)x(256,128) -> 1024x4 = 4096 waves
  gemm_f16_wmma    <<<  512, 256, 0, stream>>>(yh, WoutT, opre, MROWS, D_MODELC, 256);
  // group-norm + SiLU + residual
  gn_stats_kernel  <<<   32, 256, 0, stream>>>(opre, stats);
  finalize_kernel  <<<16384, 256, 0, stream>>>(opre, x_hsi, stats, gamma, beta, out);
}
